// ODEFunc_4758823764209
// MI455X (gfx1250) — compile-verified, matrix operand order checked by reading, not run on hardware
//
#include <hip/hip_runtime.h>
#include <hip/hip_bf16.h>

typedef __attribute__((ext_vector_type(16))) _Float16 v16h;
typedef __attribute__((ext_vector_type(8)))  _Float16 v8h;
typedef __attribute__((ext_vector_type(8)))  float    v8f;

#define BATCH  16384
#define DIM    512
#define HIDDEN 2048

// GEMM tile config: block = 256 threads = 8 waves (wave32)
// block tile 128x128, K step 32; wave tile 32(M) x 64(N) -> 2x4 WMMA tiles
#define BM 128
#define BN 128
#define BKK 32

enum { EPI_TANH = 0, EPI_TANH_D2 = 1, EPI_BIAS_OUT = 2, EPI_MUL_DTANH = 3, EPI_STORE_F32 = 4 };

// Load a 16x32 f16 WMMA operand fragment for this lane from an LDS tile with
// row stride BKK halfs. Per CDNA5 ISA 16-bit A layout:
//   lanes 0-15 : row = lane,    K = 0..7  (v0..3), K = 16..23 (v4..7)
//   lanes 16-31: row = lane-16, K = 8..15 (v0..3), K = 24..31 (v4..7)
__device__ __forceinline__ v16h ldfrag(const _Float16* p) {
  v8h lo = *(const v8h*)(p);
  v8h hi = *(const v8h*)(p + 16);
  v16h r;
#pragma unroll
  for (int i = 0; i < 8; ++i) { r[i] = lo[i]; r[i + 8] = hi[i]; }
  return r;
}

// C[M,N] = A[M,K] @ W[N,K]^T  (+ fused epilogue).  M%128==0, N%128==0, K%32==0.
template <int EPI>
__global__ __launch_bounds__(256)
void gemm_wmma_k(const _Float16* __restrict__ A,
                 const _Float16* __restrict__ W,
                 const float* __restrict__ bias,   // [N]  (EPI 0,1,2)
                 const float* __restrict__ s3v,    // [N]  (EPI 1)
                 const _Float16* __restrict__ H1,  // [M,N] (EPI 3)
                 _Float16* __restrict__ Hout,      // f16 out, ld = N (EPI 0,1)
                 _Float16* __restrict__ Dout,      // f16 out, ld = N (EPI 1,3)
                 float* __restrict__ Fout,         // f32 out, ld = ldF (EPI 2,4)
                 int M, int N, int K, int ldF) {
  __shared__ _Float16 As[BM * BKK];
  __shared__ _Float16 Bs[BN * BKK];

  const int tid  = threadIdx.x;
  const int lane = tid & 31;
  const int wave = tid >> 5;
  const int wm   = wave >> 1;  // 0..3 -> M offset wm*32
  const int wn   = wave & 1;   // 0..1 -> N offset wn*64
  const long bm  = (long)blockIdx.y * BM;
  const long bn  = (long)blockIdx.x * BN;

  v8f acc[2][4] = {};

  const int lr = lane & 15;
  const int kh = (lane < 16) ? 0 : 8;

  for (int k0 = 0; k0 < K; k0 += BKK) {
    // Stage 128x32-half tiles of A and of W (rows of W == B^T rows).
#pragma unroll
    for (int i = 0; i < 2; ++i) {
      int v = tid + i * 256;          // 512 v8h vectors per tile
      int r = v >> 2;
      int c = (v & 3) * 8;
      *(v8h*)(&As[r * BKK + c]) = *(const v8h*)(&A[(bm + r) * (long)K + k0 + c]);
      *(v8h*)(&Bs[r * BKK + c]) = *(const v8h*)(&W[(bn + r) * (long)K + k0 + c]);
    }
    __syncthreads();

    v16h af[2], bf[4];
#pragma unroll
    for (int mi = 0; mi < 2; ++mi)
      af[mi] = ldfrag(&As[(wm * 32 + mi * 16 + lr) * BKK + kh]);
#pragma unroll
    for (int ni = 0; ni < 4; ++ni)
      bf[ni] = ldfrag(&Bs[(wn * 64 + ni * 16 + lr) * BKK + kh]);

#pragma unroll
    for (int mi = 0; mi < 2; ++mi)
#pragma unroll
      for (int ni = 0; ni < 4; ++ni)
        acc[mi][ni] = __builtin_amdgcn_wmma_f32_16x16x32_f16(
            false, af[mi], false, bf[ni], (short)0, acc[mi][ni], false, false);
    __syncthreads();
  }

  // Epilogue. C/D layout: VGPR r -> M = r (lanes 0-15) or r+8 (lanes 16-31),
  // N = lane&15 within the 16x16 tile.
  const int rofs = (lane >= 16) ? 8 : 0;
#pragma unroll
  for (int mi = 0; mi < 2; ++mi) {
#pragma unroll
    for (int ni = 0; ni < 4; ++ni) {
      const long gcol = bn + wn * 64 + ni * 16 + lr;
      float bval = 0.0f, sval = 0.0f;
      if (EPI == EPI_TANH || EPI == EPI_TANH_D2 || EPI == EPI_BIAS_OUT) bval = bias[gcol];
      if (EPI == EPI_TANH_D2) sval = s3v[gcol];
#pragma unroll
      for (int r = 0; r < 8; ++r) {
        const long grow = bm + wm * 32 + mi * 16 + r + rofs;
        float v = acc[mi][ni][r] + bval;
        if (EPI == EPI_TANH) {
          Hout[grow * N + gcol] = (_Float16)tanhf(v);
        } else if (EPI == EPI_TANH_D2) {
          float h = tanhf(v);
          Hout[grow * N + gcol] = (_Float16)h;
          Dout[grow * N + gcol] = (_Float16)(sval * (1.0f - h * h));
        } else if (EPI == EPI_BIAS_OUT) {
          Fout[grow * (long)ldF + gcol] = v;
        } else if (EPI == EPI_MUL_DTANH) {
          float h = (float)H1[grow * N + gcol];
          Dout[grow * N + gcol] = (_Float16)(v * (1.0f - h * h));
        } else {  // EPI_STORE_F32
          Fout[grow * (long)ldF + gcol] = v;
        }
      }
    }
  }
}

// z (f16 copy of aug[:, :512]) and esum = sum_k eps[k]
__global__ void prep_z_esum_k(const float* __restrict__ aug,
                              const float* __restrict__ eps,
                              _Float16* __restrict__ zh,
                              float* __restrict__ esum) {
  long i = (long)blockIdx.x * blockDim.x + threadIdx.x;
  if (i >= (long)BATCH * DIM) return;
  long b = i >> 9;          // /512
  long d = i & (DIM - 1);
  zh[i] = (_Float16)aug[b * (DIM + 1) + d];
  const long S = (long)BATCH * DIM;
  esum[i] = eps[i] + eps[i + S] + eps[i + 2 * S];
}

__global__ void conv_h_k(const float* __restrict__ src, _Float16* __restrict__ dst, long n) {
  long i = (long)blockIdx.x * blockDim.x + threadIdx.x;
  if (i < n) dst[i] = (_Float16)src[i];
}

// dst[C,R] (f16) = transpose(src[R,C] f32): dst[c*R + r] = src[r*C + c]
__global__ void trans_h_k(const float* __restrict__ src, _Float16* __restrict__ dst,
                          int R, int C) {
  long i = (long)blockIdx.x * blockDim.x + threadIdx.x;
  long n = (long)R * C;
  if (i >= n) return;
  long c = i / R;
  long r = i - c * R;
  dst[i] = (_Float16)src[r * (long)C + c];
}

// s3[j] = sum_i W3[i,j]   (W3: 512 x 2048)
__global__ void s3_k(const float* __restrict__ W3, float* __restrict__ s3) {
  int j = blockIdx.x * blockDim.x + threadIdx.x;
  if (j >= HIDDEN) return;
  float s = 0.0f;
  for (int i = 0; i < DIM; ++i) s += W3[(long)i * HIDDEN + j];
  s3[j] = s;
}

// trace[b] = dot(esum[b,:], g[b,:]) / 3 -> d_out[b*513 + 512]
__global__ void trace_k(const float* __restrict__ esum, const float* __restrict__ g,
                        float* __restrict__ out) {
  __shared__ float red[256];
  long b = blockIdx.x;
  float s = 0.0f;
  for (int d = threadIdx.x; d < DIM; d += 256)
    s += esum[b * DIM + d] * g[b * DIM + d];
  red[threadIdx.x] = s;
  __syncthreads();
  for (int off = 128; off > 0; off >>= 1) {
    if (threadIdx.x < off) red[threadIdx.x] += red[threadIdx.x + off];
    __syncthreads();
  }
  if (threadIdx.x == 0) out[b * (long)(DIM + 1) + DIM] = red[0] * (1.0f / 3.0f);
}

extern "C" void kernel_launch(void* const* d_in, const int* in_sizes, int n_in,
                              void* d_out, int out_size, void* d_ws, size_t ws_size,
                              hipStream_t stream) {
  const float* aug = (const float*)d_in[1];  // [16384, 513]
  const float* eps = (const float*)d_in[2];  // [3, 16384, 512]
  const float* W1  = (const float*)d_in[3];  // [2048, 512]
  const float* b1  = (const float*)d_in[4];
  const float* W2  = (const float*)d_in[5];  // [2048, 2048]
  const float* b2  = (const float*)d_in[6];
  const float* W3  = (const float*)d_in[7];  // [512, 2048]
  const float* b3  = (const float*)d_in[8];
  float* out = (float*)d_out;                // [16384, 513]

  char* ws = (char*)d_ws;
  size_t off = 0;
  auto alloc = [&](size_t bytes) { char* p = ws + off; off += (bytes + 255) & ~(size_t)255; return p; };

  _Float16* zh   = (_Float16*)alloc((size_t)BATCH * DIM * 2);       // 16 MB
  float*    esum = (float*)   alloc((size_t)BATCH * DIM * 4);       // 32 MB
  _Float16* W1h  = (_Float16*)alloc((size_t)HIDDEN * DIM * 2);      //  2 MB
  _Float16* W2h  = (_Float16*)alloc((size_t)HIDDEN * HIDDEN * 2);   //  8 MB
  _Float16* W3h  = (_Float16*)alloc((size_t)DIM * HIDDEN * 2);      //  2 MB
  _Float16* W2Th = (_Float16*)alloc((size_t)HIDDEN * HIDDEN * 2);   //  8 MB
  _Float16* W1Th = (_Float16*)alloc((size_t)DIM * HIDDEN * 2);      //  2 MB
  float*    s3v  = (float*)   alloc((size_t)HIDDEN * 4);
  _Float16* h1h  = (_Float16*)alloc((size_t)BATCH * HIDDEN * 2);    // 64 MB
  _Float16* h2h  = (_Float16*)alloc((size_t)BATCH * HIDDEN * 2);    // 64 MB
  _Float16* d2h  = (_Float16*)alloc((size_t)BATCH * HIDDEN * 2);    // 64 MB
  _Float16* d1h  = h2h;           // reuse: h2 dead after forward GEMM3
  float*    g    = (float*)d2h;   // reuse: d2 dead after backward GEMM4
  (void)ws_size; (void)in_sizes; (void)n_in; (void)out_size;

  // --- precompute ---
  {
    long n = (long)BATCH * DIM;
    prep_z_esum_k<<<dim3((n + 255) / 256), 256, 0, stream>>>(aug, eps, zh, esum);
  }
  conv_h_k<<<dim3(((long)HIDDEN * DIM + 255) / 256), 256, 0, stream>>>(W1, W1h, (long)HIDDEN * DIM);
  conv_h_k<<<dim3(((long)HIDDEN * HIDDEN + 255) / 256), 256, 0, stream>>>(W2, W2h, (long)HIDDEN * HIDDEN);
  conv_h_k<<<dim3(((long)DIM * HIDDEN + 255) / 256), 256, 0, stream>>>(W3, W3h, (long)DIM * HIDDEN);
  trans_h_k<<<dim3(((long)HIDDEN * HIDDEN + 255) / 256), 256, 0, stream>>>(W2, W2Th, HIDDEN, HIDDEN);
  trans_h_k<<<dim3(((long)HIDDEN * DIM + 255) / 256), 256, 0, stream>>>(W1, W1Th, HIDDEN, DIM);
  s3_k<<<dim3(HIDDEN / 256), 256, 0, stream>>>(W3, s3v);

  const dim3 blk(256);
  const dim3 gH(HIDDEN / BN, BATCH / BM);  // N = 2048
  const dim3 gD(DIM / BN, BATCH / BM);     // N = 512

  // --- forward ---
  // h1 = tanh(z @ W1^T + b1)
  gemm_wmma_k<EPI_TANH><<<gH, blk, 0, stream>>>(zh, W1h, b1, nullptr, nullptr,
      h1h, nullptr, nullptr, BATCH, HIDDEN, DIM, 0);
  // h2 = tanh(h1 @ W2^T + b2); d2 = s3 * (1 - h2^2)
  gemm_wmma_k<EPI_TANH_D2><<<gH, blk, 0, stream>>>(h1h, W2h, b2, s3v, nullptr,
      h2h, d2h, nullptr, BATCH, HIDDEN, HIDDEN, 0);
  // dz_dt = h2 @ W3^T + b3 -> d_out[:, 0:512] (row stride 513)
  gemm_wmma_k<EPI_BIAS_OUT><<<gD, blk, 0, stream>>>(h2h, W3h, b3, nullptr, nullptr,
      nullptr, nullptr, out, BATCH, DIM, HIDDEN, DIM + 1);

  // --- backward (cotangent = ones) ---
  // d1 = (d2 @ W2) * (1 - h1^2)   [W2Th is W2^T so TRANS-GEMM gives d2 @ W2]
  gemm_wmma_k<EPI_MUL_DTANH><<<gH, blk, 0, stream>>>(d2h, W2Th, nullptr, nullptr, h1h,
      nullptr, d1h, nullptr, BATCH, HIDDEN, HIDDEN, 0);
  // g = d1 @ W1
  gemm_wmma_k<EPI_STORE_F32><<<gD, blk, 0, stream>>>(d1h, W1Th, nullptr, nullptr, nullptr,
      nullptr, nullptr, g, BATCH, DIM, HIDDEN, DIM);

  // trace -> d_out[:, 512]
  trace_k<<<dim3(BATCH), 256, 0, stream>>>(esum, g, out);
}
